// MPSWA_81819126989059
// MI455X (gfx1250) — matrix operand order repaired and checked by reading, not compiled, not run on hardware
//
#include <hip/hip_runtime.h>

typedef __attribute__((ext_vector_type(16))) __bf16 v16bf;
typedef __attribute__((ext_vector_type(8)))  float  v8f;

union BF16x16 { v16bf v; uint4 q[2]; };

constexpr int B_ = 4, S_ = 2048, C_ = 88, HEAD = 4, DK = 22, WD = 14;
constexpr int NTOK = B_ * S_;            // 8192 tokens
constexpr int C3 = 3 * C_;               // 264
constexpr float BN_INV = 0.99999500003749969f;   // 1/sqrt(1+1e-5)
constexpr float QSCALE = 0.21320071635561041f;   // 1/sqrt(22)

__constant__ float DEC_LO[8] = {
  -0.010597401784997278f, 0.032883011666982945f, 0.030841381835986965f,
  -0.18703481171888114f, -0.02798376941698385f, 0.6308807679295904f,
   0.7148465705525415f,   0.23037781330885523f };
__constant__ float DEC_HI[8] = {
  -0.23037781330885523f,  0.7148465705525415f,  -0.6308807679295904f,
  -0.02798376941698385f,  0.18703481171888114f,  0.030841381835986965f,
  -0.032883011666982945f, -0.010597401784997278f };

// ---------------- fused 1x1 convs (branch1 final, branch2/3 stems) ----------
__global__ void k_pw3(const float* __restrict__ x,
                      const float* __restrict__ w1,  const float* __restrict__ g1,  const float* __restrict__ b1,
                      const float* __restrict__ w2a, const float* __restrict__ g2a, const float* __restrict__ b2a,
                      const float* __restrict__ w3a, const float* __restrict__ g3a, const float* __restrict__ b3a,
                      float* __restrict__ x1, float* __restrict__ t2a, float* __restrict__ t3a) {
  __shared__ float xr[C_];
  const int tok = blockIdx.x;
  const int t = threadIdx.x;
  if (t < C_) xr[t] = x[(size_t)tok * C_ + t];
  __syncthreads();
  if (t < C_) {
    float a1 = 0.f, a2 = 0.f, a3 = 0.f;
    const float* r1 = w1  + t * C_;
    const float* r2 = w2a + t * C_;
    const float* r3 = w3a + t * C_;
    #pragma unroll 8
    for (int i = 0; i < C_; ++i) {
      float xv = xr[i];
      a1 = fmaf(xv, r1[i], a1);
      a2 = fmaf(xv, r2[i], a2);
      a3 = fmaf(xv, r3[i], a3);
    }
    x1 [(size_t)tok*C_+t] = fmaxf(fmaf(g1[t]*BN_INV,  a1, b1[t]),  0.f);
    t2a[(size_t)tok*C_+t] = fmaxf(fmaf(g2a[t]*BN_INV, a2, b2a[t]), 0.f);
    t3a[(size_t)tok*C_+t] = fmaxf(fmaf(g3a[t]*BN_INV, a3, b3a[t]), 0.f);
  }
}

// ---------------- odd-tap conv along S (zero 'same' padding) ----------------
template<int TAPS>
__global__ void k_conv(const float* __restrict__ in, const float* __restrict__ w,
                       const float* __restrict__ g,  const float* __restrict__ bb,
                       float* __restrict__ out) {
  __shared__ float xr[TAPS * C_];
  const int tok = blockIdx.x;
  const int b = tok / S_, s = tok % S_;
  const int P = (TAPS - 1) / 2;
  for (int j = threadIdx.x; j < TAPS * C_; j += blockDim.x) {
    int r = j / C_, i = j % C_;
    int ss = s + r - P;
    xr[j] = (ss >= 0 && ss < S_) ? in[((size_t)b * S_ + ss) * C_ + i] : 0.f;
  }
  __syncthreads();
  const int t = threadIdx.x;
  if (t < C_) {
    float acc = 0.f;
    const float* wr = w + (size_t)t * C_ * TAPS;
    for (int i = 0; i < C_; ++i) {
      #pragma unroll
      for (int tap = 0; tap < TAPS; ++tap)
        acc = fmaf(xr[tap * C_ + i], wr[i * TAPS + tap], acc);
    }
    out[(size_t)tok * C_ + t] = fmaxf(fmaf(g[t] * BN_INV, acc, bb[t]), 0.f);
  }
}

// ---------------- 3C -> C merge + residual -> mp ----------------------------
__global__ void k_mp(const float* __restrict__ x1, const float* __restrict__ x2, const float* __restrict__ x3,
                     const float* __restrict__ Wmp, const float* __restrict__ bmp,
                     const float* __restrict__ x, float* __restrict__ mp) {
  __shared__ float cat[C3];
  const int tok = blockIdx.x;
  for (int j = threadIdx.x; j < C_; j += blockDim.x) {
    cat[j]        = x1[(size_t)tok*C_+j];
    cat[C_ + j]   = x2[(size_t)tok*C_+j];
    cat[2*C_ + j] = x3[(size_t)tok*C_+j];
  }
  __syncthreads();
  const int t = threadIdx.x;
  if (t < C_) {
    float acc = bmp[t] + x[(size_t)tok*C_+t];
    const float* wr = Wmp + (size_t)t * C3;
    #pragma unroll 8
    for (int cc = 0; cc < C3; ++cc) acc = fmaf(cat[cc], wr[cc], acc);
    mp[(size_t)tok*C_+t] = acc;
  }
}

// -------- QKV projection + db4 DWT of V + bf16 WMMA-layout packing ----------
// qbf,kbf: [b][h][s][32] bf16, dk 22 zero-padded to 32 (fragment friendly)
// cdt:     [b][h][d(16)][s] bf16, wd 14 zero-padded to 16 (cD^T, key-contig)
__global__ void k_qkv(const float* __restrict__ mp,
                      const float* __restrict__ Wq, const float* __restrict__ Wk, const float* __restrict__ Wv,
                      __bf16* __restrict__ qbf, __bf16* __restrict__ kbf,
                      float* __restrict__ cA, __bf16* __restrict__ cdt) {
  __shared__ float mr[C_], qr[C_], kr[C_], vr[C_], cdr[HEAD * WD];
  const int tok = blockIdx.x;
  const int b = tok / S_, s = tok % S_;
  const int t = threadIdx.x;
  if (t < C_) mr[t] = mp[(size_t)tok*C_+t];
  __syncthreads();
  if (t < C_) {
    float aq = 0.f, ak = 0.f, av = 0.f;
    const float* rq = Wq + t*C_; const float* rk = Wk + t*C_; const float* rv = Wv + t*C_;
    #pragma unroll 8
    for (int i = 0; i < C_; ++i) {
      float m = mr[i];
      aq = fmaf(m, rq[i], aq); ak = fmaf(m, rk[i], ak); av = fmaf(m, rv[i], av);
    }
    qr[t] = aq * QSCALE;   // fold dk^-0.5 into Q
    kr[t] = ak;
    vr[t] = av;
  }
  __syncthreads();
  if (t < HEAD * WD) {
    int h = t / WD, d = t % WD;
    const float* vh = vr + h * DK;
    float lo = 0.f, hi = 0.f;
    #pragma unroll
    for (int gi = 0; gi < 8; ++gi) {
      int idx = 2*d + 1 - gi;                 // pywt symmetric boundary
      if (idx < 0) idx = -1 - idx;
      if (idx >= DK) idx = 2*DK - 1 - idx;
      float xv = vh[idx];
      lo = fmaf(DEC_LO[gi], xv, lo);
      hi = fmaf(DEC_HI[gi], xv, hi);
    }
    cA[(size_t)tok*(HEAD*WD)+t] = lo;
    cdr[t] = hi;
  }
  __syncthreads();
  for (int j = t; j < HEAD * 32; j += blockDim.x) {
    int h = j >> 5, kk = j & 31;
    size_t off = ((size_t)(b*HEAD + h) * S_ + s) * 32 + kk;
    qbf[off] = (__bf16)(kk < DK ? qr[h*DK + kk] : 0.f);
    kbf[off] = (__bf16)(kk < DK ? kr[h*DK + kk] : 0.f);
  }
  for (int j = t; j < HEAD * 16; j += blockDim.x) {
    int h = j >> 4, d = j & 15;
    cdt[((size_t)(b*HEAD + h) * 16 + d) * S_ + s] = (__bf16)(d < WD ? cdr[h*WD + d] : 0.f);
  }
}

// ------- flash attention, transposed-score formulation (bf16 WMMA) ----------
// S^T = K_tile(A) x Q^T(B)  -> per-lane softmax stats (query = lane&15)
// O^T = cD^T(A) x P^T(B)    -> P^T built from score frags via lane^16 swap
__global__ void __launch_bounds__(128)
k_attn(const __bf16* __restrict__ qp, const __bf16* __restrict__ kp,
       const __bf16* __restrict__ cdt, float* __restrict__ wcd) {
  const int wave = threadIdx.x >> 5;
  const int lane = threadIdx.x & 31;
  const int hf   = lane >> 4;              // wave32 half
  const int nn   = lane & 15;              // query column for this lane
  const int bh = blockIdx.x / (S_ / 64);   // b*HEAD + h
  const int qt = blockIdx.x % (S_ / 64);
  const int qbase = qt * 64 + wave * 16;

  // Q^T as B-fragment: lane -> query row qbase+nn, dims 8*hf..8*hf+7 and 16+8*hf..16+8*hf+7
  BF16x16 qf;
  {
    const __bf16* qrow = qp + ((size_t)bh * S_ + qbase + nn) * 32 + 8 * hf;
    qf.q[0] = *(const uint4*)(qrow);
    qf.q[1] = *(const uint4*)(qrow + 16);
  }

  float mrun = -3.0e38f, lrun = 0.f;
  v8f o = {0.f,0.f,0.f,0.f,0.f,0.f,0.f,0.f};

  for (int kt = 0; kt < S_ / 64; ++kt) {
    const int kb64 = kt * 64;
    // 4 transposed score tiles: A = K rows (16 keys x 32 dims)
    v8f sc[4];
    #pragma unroll
    for (int j = 0; j < 4; ++j) {
      BF16x16 ka;
      const __bf16* krow = kp + ((size_t)bh * S_ + kb64 + 16*j + nn) * 32;
      ka.q[0] = *(const uint4*)(krow + 8 * hf);
      ka.q[1] = *(const uint4*)(krow + 16 + 8 * hf);
      v8f c0 = {0.f,0.f,0.f,0.f,0.f,0.f,0.f,0.f};
      sc[j] = __builtin_amdgcn_wmma_f32_16x16x32_bf16(false, ka.v, false, qf.v,
                                                      (short)0, c0, false, false);
    }
    if (kt + 1 < S_ / 64)                  // global_prefetch next K block
      __builtin_prefetch(kp + ((size_t)bh * S_ + kb64 + 64 + nn) * 32, 0, 0);

    // per-lane online softmax over this lane's 32 keys + partner's 32
    float mloc = sc[0][0];
    #pragma unroll
    for (int j = 0; j < 4; ++j)
      #pragma unroll
      for (int r = 0; r < 8; ++r) mloc = fmaxf(mloc, sc[j][r]);
    float mnew = fmaxf(mrun, fmaxf(mloc, __shfl_xor(mloc, 16, 32)));
    float alpha = __expf(mrun - mnew);
    mrun = mnew;

    float pv[4][8];
    float sloc = 0.f;
    #pragma unroll
    for (int j = 0; j < 4; ++j)
      #pragma unroll
      for (int r = 0; r < 8; ++r) {
        float p = __expf(sc[j][r] - mnew);
        pv[j][r] = p;
        sloc += p;
      }
    lrun = lrun * alpha + (sloc + __shfl_xor(sloc, 16, 32));

    // build P^T B-fragments (keys 0..31 and 32..63) with one half-swap
    v16bf pb0, pb1;
    #pragma unroll
    for (int g = 0; g < 2; ++g) {
      float ex[8];
      #pragma unroll
      for (int r = 0; r < 8; ++r)
        ex[r] = __shfl_xor(hf ? pv[2*g][r] : pv[2*g + 1][r], 16, 32);
      #pragma unroll
      for (int i = 0; i < 8; ++i) {
        __bf16 lo = (__bf16)(hf ? ex[i] : pv[2*g][i]);       // keys 32g+16*hf+{0..7}
        __bf16 hi = (__bf16)(hf ? pv[2*g + 1][i] : ex[i]);   // keys 32g+16*hf+{8..15}
        if (g == 0) { pb0[i] = lo; pb0[8 + i] = hi; }
        else        { pb1[i] = lo; pb1[8 + i] = hi; }
      }
    }

    // cD^T A-fragments: rows d = nn, keys contiguous in cdt
    BF16x16 da0, da1;
    {
      const __bf16* drow = cdt + ((size_t)bh * 16 + nn) * S_ + kb64;
      da0.q[0] = *(const uint4*)(drow + 16 * hf);
      da0.q[1] = *(const uint4*)(drow + 8 + 16 * hf);
      da1.q[0] = *(const uint4*)(drow + 32 + 16 * hf);
      da1.q[1] = *(const uint4*)(drow + 40 + 16 * hf);
    }

    #pragma unroll
    for (int r = 0; r < 8; ++r) o[r] *= alpha;   // scalar alpha (per query)
    o = __builtin_amdgcn_wmma_f32_16x16x32_bf16(false, da0.v, false, pb0,
                                                (short)0, o, false, false);
    o = __builtin_amdgcn_wmma_f32_16x16x32_bf16(false, da1.v, false, pb1,
                                                (short)0, o, false, false);
  }

  // O^T: lane holds d = r + 8*hf for query qbase+nn
  {
    float inv = 1.f / lrun;
    #pragma unroll
    for (int r = 0; r < 8; ++r) {
      int d = r + (hf << 3);
      if (d < WD)
        wcd[((size_t)bh * S_ + qbase + nn) * WD + d] = o[r] * inv;
    }
  }
}

// ---------------- output projection + residual ------------------------------
// wcd layout: [b][h][s][wd]; xo flatten wants [s][h][wd]
__global__ void k_out(const float* __restrict__ wcd, const float* __restrict__ cA,
                      const float* __restrict__ Wout, const float* __restrict__ bout,
                      const float* __restrict__ mp, float* __restrict__ out) {
  __shared__ float cat[2 * HEAD * WD];     // 112
  const int tok = blockIdx.x;
  const int b = tok / S_, s = tok % S_;
  const int t = threadIdx.x;
  if (t < HEAD * WD) {
    int h = t / WD, d = t % WD;
    cat[t]           = fmaxf(wcd[(((size_t)b * HEAD + h) * S_ + s) * WD + d], 0.f);
    cat[HEAD*WD + t] = cA[(size_t)tok*(HEAD*WD) + t];
  }
  __syncthreads();
  if (t < C_) {
    float acc = bout[t];
    const float* wr = Wout + (size_t)t * (2 * HEAD * WD);
    #pragma unroll 8
    for (int j = 0; j < 2 * HEAD * WD; ++j) acc = fmaf(cat[j], wr[j], acc);
    out[(size_t)tok*C_ + t] = mp[(size_t)tok*C_ + t] + acc;
  }
}

// ---------------- launcher ---------------------------------------------------
extern "C" void kernel_launch(void* const* d_in, const int* in_sizes, int n_in,
                              void* d_out, int out_size, void* d_ws, size_t ws_size,
                              hipStream_t stream) {
  (void)in_sizes; (void)n_in; (void)out_size; (void)ws_size;
  const float* x    = (const float*)d_in[0];
  const float* w1   = (const float*)d_in[1];
  const float* w2a  = (const float*)d_in[2];
  const float* w2b  = (const float*)d_in[3];
  const float* w3a  = (const float*)d_in[4];
  const float* w3b  = (const float*)d_in[5];
  const float* w3c  = (const float*)d_in[6];
  const float* g1   = (const float*)d_in[7];
  const float* g2a  = (const float*)d_in[8];
  const float* g2b  = (const float*)d_in[9];
  const float* g3a  = (const float*)d_in[10];
  const float* g3b  = (const float*)d_in[11];
  const float* g3c  = (const float*)d_in[12];
  const float* b1   = (const float*)d_in[13];
  const float* b2a  = (const float*)d_in[14];
  const float* b2b  = (const float*)d_in[15];
  const float* b3a  = (const float*)d_in[16];
  const float* b3b  = (const float*)d_in[17];
  const float* b3c  = (const float*)d_in[18];
  const float* Wmp  = (const float*)d_in[19];
  const float* bmp  = (const float*)d_in[20];
  const float* Wq   = (const float*)d_in[21];
  const float* Wk   = (const float*)d_in[22];
  const float* Wv   = (const float*)d_in[23];
  const float* Wout = (const float*)d_in[24];
  const float* bout = (const float*)d_in[25];

  float* ws = (float*)d_ws;
  const size_t N = (size_t)NTOK * C_;        // 720896
  const size_t M = (size_t)NTOK * HEAD * WD; // 458752
  float* x1  = ws;
  float* t2a = ws + 1*N;
  float* x2  = ws + 2*N;
  float* t3a = ws + 3*N;
  float* t3b = ws + 4*N;
  float* x3  = ws + 5*N;
  float* mp  = ws + 6*N;
  float* cA  = ws + 7*N;
  float* wcd = cA + M;
  __bf16* qbf = (__bf16*)(wcd + M);
  __bf16* kbf = qbf + (size_t)B_ * HEAD * S_ * 32;
  __bf16* cdt = kbf + (size_t)B_ * HEAD * S_ * 32;

  dim3 tb(96);
  k_pw3<<<NTOK, tb, 0, stream>>>(x, w1,g1,b1, w2a,g2a,b2a, w3a,g3a,b3a, x1, t2a, t3a);
  k_conv<3><<<NTOK, tb, 0, stream>>>(t2a, w2b, g2b, b2b, x2);
  k_conv<3><<<NTOK, tb, 0, stream>>>(t3a, w3b, g3b, b3b, t3b);
  k_conv<5><<<NTOK, tb, 0, stream>>>(t3b, w3c, g3c, b3c, x3);
  k_mp <<<NTOK, tb, 0, stream>>>(x1, x2, x3, Wmp, bmp, x, mp);
  k_qkv<<<NTOK, tb, 0, stream>>>(mp, Wq, Wk, Wv, qbf, kbf, cA, cdt);
  k_attn<<<B_ * HEAD * (S_ / 64), 128, 0, stream>>>(qbf, kbf, cdt, wcd);
  k_out<<<NTOK, 128, 0, stream>>>(wcd, cA, Wout, bout, mp, (float*)d_out);
}
